// Attention3_18915035971820
// MI455X (gfx1250) — matrix operand order repaired and checked by reading, not compiled, not run on hardware
//
#include <hip/hip_runtime.h>
#include <hip/hip_bf16.h>
#include <math.h>

#define B_ 32
#define L_ 2048
#define D_ 1024

typedef __attribute__((ext_vector_type(16))) __bf16 v16bf;
typedef __attribute__((ext_vector_type(8)))  float  v8f;

union BfFrag { v16bf v; uint4 q[2]; };

static __device__ inline unsigned pack2_bf16(float lo, float hi) {
    __bf16 l = (__bf16)lo, h = (__bf16)hi;
    unsigned short ul, uh;
    __builtin_memcpy(&ul, &l, 2);
    __builtin_memcpy(&uh, &h, 2);
    return ((unsigned)uh << 16) | (unsigned)ul;
}

// ---------------- q = decoder_hidden @ W_w.T + W_b   [B, D] ----------------
__global__ void qproj_kernel(const float* __restrict__ dh, const float* __restrict__ Ww,
                             const float* __restrict__ Wb, float* __restrict__ q) {
    int idx = blockIdx.x * blockDim.x + threadIdx.x;   // B_*D_ threads
    int b = idx >> 10, n = idx & (D_ - 1);
    const float4* x = (const float4*)(dh + (size_t)b * D_);
    const float4* w = (const float4*)(Ww + (size_t)n * D_);
    float acc = Wb[n];
    #pragma unroll 4
    for (int i = 0; i < D_ / 4; ++i) {
        float4 a = x[i], c = w[i];
        acc += a.x * c.x + a.y * c.y + a.z * c.z + a.w * c.w;
    }
    q[idx] = acc;
}

// ---------------- convert U_w (f32) -> bf16 copy in workspace ----------------
__global__ void cvt_kernel(const float* __restrict__ U, unsigned* __restrict__ ubf) {
    int idx = blockIdx.x * blockDim.x + threadIdx.x;   // D_*D_/2 threads
    float2 f = ((const float2*)U)[idx];
    ubf[idx] = pack2_bf16(f.x, f.y);
}

// ---------------- fused  scores = v . tanh(q + enc @ U_w.T + U_b) + v_b ------
// grid: (L/32, B), 256 threads = 8 wave32s. Each block: 32 encoder rows
// (two 16-row M-subtiles, so each B fragment feeds two WMMAs).
// Wave w owns output columns [w*128, w*128+128).
__global__ __launch_bounds__(256)
void score_kernel(const float* __restrict__ enc,
                  const unsigned short* __restrict__ Ubf,
                  const float* __restrict__ q,
                  const float* __restrict__ Ub,
                  const float* __restrict__ vw,
                  const float* __restrict__ vb,
                  float* __restrict__ scores) {
    __shared__ __align__(16) unsigned short ldsA[32][1032]; // bf16, +8 pad vs banks
    __shared__ float ldsRed[8 * 32];

    const int tid   = threadIdx.x;
    const int lane  = tid & 31;
    const int w     = tid >> 5;
    const int b     = blockIdx.y;
    const int Mbase = blockIdx.x * 32;

    const float* encT = enc + ((size_t)b * L_ + Mbase) * D_;

    // Stage A-tile (32 x 1024) into LDS as packed bf16 pairs.
    const float2* e2 = (const float2*)encT;
    #pragma unroll 8
    for (int i = 0; i < 64; ++i) {
        int p = tid + i * 256;            // 16384 float2 pairs total
        int row = p >> 9, c2 = p & 511;
        float2 f = e2[p];
        *((unsigned*)&ldsA[row][0] + c2) = pack2_bf16(f.x, f.y);
    }
    __syncthreads();

    const int m    = lane & 15;           // A row (within subtile) / B & C column
    const int half = lane >> 4;
    const int khA  = half * 8;            // A frag: K 0-7/16-23 vs 8-15/24-31
    const int khB  = half * 8;            // B frag: K 0-7/16-23 vs 8-15/24-31
    const int Nw   = w * 128;

    const v8f zero = {0.f, 0.f, 0.f, 0.f, 0.f, 0.f, 0.f, 0.f};
    v8f acc[16];                          // [msub][nt]
    #pragma unroll
    for (int i = 0; i < 16; ++i) acc[i] = zero;

    for (int kt = 0; kt < 32; ++kt) {
        const int K0 = kt * 32;
        BfFrag A0, A1;
        A0.q[0] = *(const uint4*)&ldsA[m][K0 + khA];
        A0.q[1] = *(const uint4*)&ldsA[m][K0 + khA + 16];
        A1.q[0] = *(const uint4*)&ldsA[m + 16][K0 + khA];
        A1.q[1] = *(const uint4*)&ldsA[m + 16][K0 + khA + 16];
        #pragma unroll
        for (int nt = 0; nt < 8; ++nt) {
            int nrow = Nw + nt * 16 + m;  // row of U_w == output dim
            const uint4* bp = (const uint4*)(Ubf + (size_t)nrow * D_ + K0 + khB);
            BfFrag Bf;
            Bf.q[0] = bp[0];
            Bf.q[1] = bp[2];
            acc[nt] = __builtin_amdgcn_wmma_f32_16x16x32_bf16(
                false, A0.v, false, Bf.v, (short)0, acc[nt], false, false);
            acc[8 + nt] = __builtin_amdgcn_wmma_f32_16x16x32_bf16(
                false, A1.v, false, Bf.v, (short)0, acc[8 + nt], false, false);
        }
    }

    // Epilogue: part[msub][r] = sum over this lane's columns of tanh(k+q+Ub)*v
    float part0[8], part1[8];
    #pragma unroll
    for (int r = 0; r < 8; ++r) { part0[r] = 0.f; part1[r] = 0.f; }
    #pragma unroll
    for (int nt = 0; nt < 8; ++nt) {
        int ncol = Nw + nt * 16 + m;
        float qb = q[b * D_ + ncol] + Ub[ncol];
        float vv = vw[ncol];
        #pragma unroll
        for (int r = 0; r < 8; ++r) {
            part0[r] += tanhf(acc[nt][r] + qb) * vv;
            part1[r] += tanhf(acc[8 + nt][r] + qb) * vv;
        }
    }
    // reduce across the 16 lanes of each half (columns), stays within halves
    #pragma unroll
    for (int r = 0; r < 8; ++r) {
        #pragma unroll
        for (int mask = 1; mask <= 8; mask <<= 1) {
            part0[r] += __shfl_xor(part0[r], mask, 32);
            part1[r] += __shfl_xor(part1[r], mask, 32);
        }
    }
    if (m == 0) {                         // lanes 0 / 16: rows (half*8 + r)
        int base = w * 32 + half * 8;
        #pragma unroll
        for (int r = 0; r < 8; ++r) {
            ldsRed[base + r]      = part0[r];   // rows 0..15
            ldsRed[base + 16 + r] = part1[r];   // rows 16..31
        }
    }
    __syncthreads();
    if (tid < 32) {
        float s = vb[0];
        #pragma unroll
        for (int ww = 0; ww < 8; ++ww) s += ldsRed[ww * 32 + tid];
        scores[(size_t)b * L_ + Mbase + tid] = s;
    }
}

// ---------------- softmax over L, one block per batch ----------------
__global__ void softmax_kernel(const float* __restrict__ scores, float* __restrict__ attn) {
    __shared__ float red[256];
    int b = blockIdx.x, tid = threadIdx.x;
    const float* s = scores + (size_t)b * L_;
    float v[8];
    float mx = -INFINITY;
    #pragma unroll
    for (int i = 0; i < 8; ++i) { v[i] = s[tid + i * 256]; mx = fmaxf(mx, v[i]); }
    red[tid] = mx; __syncthreads();
    for (int off = 128; off > 0; off >>= 1) {
        if (tid < off) red[tid] = fmaxf(red[tid], red[tid + off]);
        __syncthreads();
    }
    mx = red[0]; __syncthreads();
    float sum = 0.f;
    #pragma unroll
    for (int i = 0; i < 8; ++i) { v[i] = __expf(v[i] - mx); sum += v[i]; }
    red[tid] = sum; __syncthreads();
    for (int off = 128; off > 0; off >>= 1) {
        if (tid < off) red[tid] += red[tid + off];
        __syncthreads();
    }
    float inv = 1.0f / red[0];
    #pragma unroll
    for (int i = 0; i < 8; ++i) attn[(size_t)b * L_ + tid + i * 256] = v[i] * inv;
}

// ---------------- context partials: split L deterministically ----------------
__global__ void ctx_partial_kernel(const float* __restrict__ enc,
                                   const float* __restrict__ attn,
                                   float* __restrict__ part) {
    int b = blockIdx.x, dbase = blockIdx.y * 256, ls = blockIdx.z * 256;
    int tid = threadIdx.x;
    const float* e = enc + ((size_t)b * L_ + ls) * D_ + dbase + tid;
    const float* a = attn + (size_t)b * L_ + ls;
    float acc = 0.f;
    #pragma unroll 8
    for (int l = 0; l < 256; ++l) acc += a[l] * e[(size_t)l * D_];
    part[((size_t)blockIdx.z * B_ + b) * D_ + dbase + tid] = acc;
}

__global__ void ctx_reduce_kernel(const float* __restrict__ part, float* __restrict__ ctx) {
    int idx = blockIdx.x * blockDim.x + threadIdx.x;   // B_*D_
    float s = 0.f;
    #pragma unroll
    for (int z = 0; z < 8; ++z) s += part[(size_t)z * B_ * D_ + idx];
    ctx[idx] = s;
}

extern "C" void kernel_launch(void* const* d_in, const int* in_sizes, int n_in,
                              void* d_out, int out_size, void* d_ws, size_t ws_size,
                              hipStream_t stream) {
    const float* dh  = (const float*)d_in[0];
    const float* enc = (const float*)d_in[1];
    const float* Ww  = (const float*)d_in[2];
    const float* Wb  = (const float*)d_in[3];
    const float* Uw  = (const float*)d_in[4];
    const float* Ub  = (const float*)d_in[5];
    const float* vw  = (const float*)d_in[6];
    const float* vb  = (const float*)d_in[7];

    float* out  = (float*)d_out;
    float* ctx  = out;                 // [B, 1, D] -> 32768 floats
    float* attn = out + B_ * D_;       // [B, 1, L] -> 65536 floats

    float* ws       = (float*)d_ws;
    float* q        = ws;                              // B*D   = 32768 f32
    float* scores   = ws + 32768;                      // B*L   = 65536 f32
    float* ctxpart  = ws + 32768 + 65536;              // 8*B*D = 262144 f32
    unsigned short* Ubf = (unsigned short*)(ws + 32768 + 65536 + 262144); // D*D bf16

    qproj_kernel<<<dim3((B_ * D_) / 256), 256, 0, stream>>>(dh, Ww, Wb, q);
    cvt_kernel<<<dim3((D_ * D_ / 2) / 256), 256, 0, stream>>>(Uw, (unsigned*)Ubf);
    score_kernel<<<dim3(L_ / 32, B_), 256, 0, stream>>>(enc, Ubf, q, Ub, vw, vb, scores);
    softmax_kernel<<<dim3(B_), 256, 0, stream>>>(scores, attn);
    ctx_partial_kernel<<<dim3(B_, D_ / 256, 8), 256, 0, stream>>>(enc, attn, ctxpart);
    ctx_reduce_kernel<<<dim3((B_ * D_) / 256), 256, 0, stream>>>(ctxpart, ctx);
}
